// StyleMaskDecoder_83219286327513
// MI455X (gfx1250) — compile-verified, matrix operand order checked by reading, not run on hardware
//
#include <hip/hip_runtime.h>
#include <hip/hip_bf16.h>

typedef __attribute__((ext_vector_type(16))) __bf16 v16bf;
typedef __attribute__((ext_vector_type(8)))  float  v8f;

#define LRELU_SLOPE 0.2f
#define EPS_DEMOD 1e-8f

static __device__ __forceinline__ unsigned short f2bf(float f) {
    union { __bf16 h; unsigned short u; } cv; cv.h = (__bf16)f; return cv.u;
}

static __device__ __forceinline__ void wait_async0() {
    asm volatile("s_wait_asynccnt 0x0" ::: "memory");
}

// Copy 16 bytes global -> LDS through the CDNA5 async path (ASYNCcnt).
static __device__ __forceinline__ void async_cp16(unsigned lds_off, const void* g) {
    asm volatile("global_load_async_to_lds_b128 %0, %1, off"
                 :: "v"(lds_off), "v"((unsigned long long)(uintptr_t)g)
                 : "memory");
}

// K index inside a 32-wide K-step for WMMA bf16 16x16x32 fragments (A and B
// symmetric), per CDNA5 ISA 16-bit A-matrix table.
static __device__ __forceinline__ int kmap(int lane, int j) {
    return ((j >> 3) << 4) + ((lane >> 4) << 3) + (((j >> 1) & 3) << 1) + (j & 1);
}

// ---------------------------------------------------------------------------
// s = istyle @ Wstyle^T + b + 1   for both style branches. 2048 outputs.
__global__ __launch_bounds__(256)
void style_k(const float* __restrict__ istyle,
             const float* __restrict__ w1, const float* __restrict__ b1,
             const float* __restrict__ w2, const float* __restrict__ b2,
             float* __restrict__ s1, float* __restrict__ s2) {
    int gid = blockIdx.x * 256 + threadIdx.x;      // 0..2047
    int sel = gid >> 10;
    int r   = gid & 1023;
    int b = r >> 8, c = r & 255;
    const float* w  = sel ? w2 : w1;
    const float* bb = sel ? b2 : b1;
    const float* iv = istyle + b * 256;
    const float* wr = w + c * 256;
    float acc = 0.f;
    for (int i = 0; i < 256; ++i) acc += iv[i] * wr[i];
    (sel ? s2 : s1)[b * 256 + c] = acc + bb[c] + 1.0f;
}

// ---------------------------------------------------------------------------
// demod[b,o] = rsqrt( sum_{i,t} W[o,i,t]^2 * s[b,i]^2 + eps ). 2048 outputs.
__global__ __launch_bounds__(256)
void demod_k(const float* __restrict__ cw1, const float* __restrict__ cw2,
             const float* __restrict__ s1, const float* __restrict__ s2,
             float* __restrict__ d1, float* __restrict__ d2) {
    int gid = blockIdx.x * 256 + threadIdx.x;      // 0..2047
    int sel = gid >> 10;
    int r   = gid & 1023;
    int b = r >> 8, o = r & 255;
    const float* W = sel ? cw2 : cw1;
    const float* s = sel ? s2 : s1;
    float acc = 0.f;
    for (int i = 0; i < 256; ++i) {
        float sv = s[b * 256 + i];
        float ss = sv * sv;
        const float* wp = W + ((long)o * 256 + i) * 9;
        float wsq = 0.f;
        #pragma unroll
        for (int t = 0; t < 9; ++t) { float w = wp[t]; wsq += w * w; }
        acc += wsq * ss;
    }
    (sel ? d2 : d1)[b * 256 + o] = rsqrtf(acc + EPS_DEMOD);
}

// ---------------------------------------------------------------------------
// Bilinear 2x upsample (align_corners=False) 64x64 -> 128x128.
// Input NCHW f32, output NHWC bf16 [b][y][x][c] (channel-fastest so the GEMM
// K dimension is contiguous in memory).
__global__ __launch_bounds__(256)
void upsample_k(const float* __restrict__ x, unsigned short* __restrict__ xu) {
    int idx = blockIdx.x * 256 + threadIdx.x;      // 4*128*128*256
    int c  = idx & 255;
    int ox = (idx >> 8) & 127;
    int oy = (idx >> 15) & 127;
    int b  = idx >> 22;
    int jy = oy >> 1; bool ey = !(oy & 1);
    int iy0 = ey ? jy - 1 : jy;     int iy1 = ey ? jy : jy + 1;
    float wy0 = ey ? 0.25f : 0.75f; float wy1 = ey ? 0.75f : 0.25f;
    iy0 = iy0 < 0 ? 0 : iy0;        iy1 = iy1 > 63 ? 63 : iy1;
    int jx = ox >> 1; bool ex = !(ox & 1);
    int ix0 = ex ? jx - 1 : jx;     int ix1 = ex ? jx : jx + 1;
    float wx0 = ex ? 0.25f : 0.75f; float wx1 = ex ? 0.75f : 0.25f;
    ix0 = ix0 < 0 ? 0 : ix0;        ix1 = ix1 > 63 ? 63 : ix1;
    const float* src = x + ((long)b * 256 + c) * 64 * 64;
    float v = wy0 * (wx0 * src[iy0 * 64 + ix0] + wx1 * src[iy0 * 64 + ix1])
            + wy1 * (wx0 * src[iy1 * 64 + ix0] + wx1 * src[iy1 * 64 + ix1]);
    xu[idx] = f2bf(v);   // idx == ((b*128+oy)*128+ox)*256 + c
}

// ---------------------------------------------------------------------------
// Pre-swizzle weights into per-lane WMMA A-fragment layout, folding
// modulation (s) and demodulation (demod) in.
// 3x3: dst[(((b*16+mt)*8 + cc)*9 + t)*512 + lane*16 + j]
//        = W[o][ci = cc*32 + kmap(lane,j)][t] * s[b,ci] * demod[b,o]
// 1x1: dst[((mt*8 + cc)*512 + lane*16 + j] = W[o][ci]
__global__ __launch_bounds__(256)
void swizzle_k(const float* __restrict__ W, const float* __restrict__ s,
               const float* __restrict__ demod, unsigned short* __restrict__ dst,
               int is3x3) {
    long e = (long)blockIdx.x * 256 + threadIdx.x;
    int j    = (int)(e & 15);
    int lane = (int)((e >> 4) & 31);
    long f   = e >> 9;
    float val;
    if (is3x3) {
        int t  = (int)(f % 9); f /= 9;
        int cc = (int)(f & 7); f >>= 3;
        int mt = (int)(f & 15);
        int b  = (int)(f >> 4);
        int o  = mt * 16 + (lane & 15);
        int ci = cc * 32 + kmap(lane, j);
        val = W[((long)o * 256 + ci) * 9 + t] * s[b * 256 + ci] * demod[b * 256 + o];
    } else {
        int cc = (int)(f & 7);
        int mt = (int)((f >> 3) & 15);
        int o  = mt * 16 + (lane & 15);
        int ci = cc * 32 + kmap(lane, j);
        val = W[(long)o * 256 + ci];
    }
    dst[e] = f2bf(val);
}

// ---------------------------------------------------------------------------
// Implicit-GEMM modulated conv, bf16 WMMA, f32 accumulate.
// Block: 256 threads = 8 waves, tile M=64 x N=32 (waves 4x2, one 16x16 each).
// Outer loop: 8 channel chunks; each stages a [3 rows][34 cols][32 ch] halo
// slab into LDS via async global->LDS b128 copies, then runs 9 fully unrolled
// WMMA tap steps from it. FUSE adds 8 more chunks of the 1x1 skip GEMM into
// the same accumulator.
template <bool FUSE>
__global__ __launch_bounds__(256)
void conv_gemm_k(const unsigned short* __restrict__ xin,   // NHWC bf16 [4,128,128,256]
                 const unsigned short* __restrict__ wfrag, // bf16 [4,16,8,9,512]
                 const unsigned short* __restrict__ xskip, // NHWC bf16 (FUSE)
                 const unsigned short* __restrict__ w1frag,// bf16 [16,8,512] (FUSE)
                 const float* __restrict__ bias,           // [256] noise bias
                 unsigned short* __restrict__ out_bf,      // conv1 act out (NHWC)
                 float* __restrict__ out_f) {              // final out (NCHW, FUSE)
    __shared__ alignas(16) unsigned short ldsb[2][3][34][32]; // [buf][row][col][ch]

    const int tid  = threadIdx.x;
    const int lane = tid & 31;
    const int wave = tid >> 5;
    const int mw = wave & 3;          // M sub-tile 0..3
    const int nw = wave >> 2;         // N sub-tile 0..1
    const int b    = blockIdx.z;
    const int mblk = blockIdx.y;      // 64-row block of out channels
    const int y    = blockIdx.x >> 2; // output row 0..127
    const int x0   = (blockIdx.x & 3) << 5;

    v8f acc = {};

    // Stage 3x34 halo sites x 32 channels (4 x 16B chunks each) for chunk cc.
    auto stage_conv = [&](int buf, int cc) {
        const int c0 = cc << 5;
        #pragma unroll
        for (int base = 0; base < 512; base += 256) {
            int ch = base + tid;
            if (ch < 408) {
                int site = ch >> 2, q = ch & 3;
                int row = site / 34;
                int col = site - row * 34;
                int yy = y + row - 1;
                int xx = x0 + col - 1;
                unsigned loff = (unsigned)(uintptr_t)&ldsb[buf][row][col][q << 3];
                if ((unsigned)yy < 128u && (unsigned)xx < 128u) {
                    const unsigned short* g =
                        xin + (((long)(b * 128 + yy) * 128 + xx) * 256 + c0 + (q << 3));
                    async_cp16(loff, g);
                } else {
                    uint4 z = {0u, 0u, 0u, 0u};
                    *(uint4*)&ldsb[buf][row][col][q << 3] = z;   // zero halo
                }
            }
        }
    };
    // Stage 32 cols x 32 channels of the skip input (no halo, never OOB).
    auto stage_skip = [&](int buf, int cs) {
        const int c0 = cs << 5;
        if (tid < 128) {
            int col = tid >> 2, q = tid & 3;
            unsigned loff = (unsigned)(uintptr_t)&ldsb[buf][0][col][q << 3];
            const unsigned short* g =
                xskip + (((long)(b * 128 + y) * 128 + x0 + col) * 256 + c0 + (q << 3));
            async_cp16(loff, g);
        }
    };

    stage_conv(0, 0);
    wait_async0();
    __syncthreads();

    const int mtg = mblk * 4 + mw;            // global 16-row tile id (0..15)
    const int kb  = (lane >> 4) << 3;         // channel base inside chunk
    const int n   = (nw << 4) + (lane & 15);  // output column within tile

    const int steps = FUSE ? 16 : 8;
    for (int it = 0; it < steps; ++it) {
        const int buf = it & 1;
        // Kick off next slab's async copies; they target the other buffer.
        if (it + 1 < steps) {
            if (it + 1 < 8) stage_conv(buf ^ 1, it + 1);
            else if (FUSE)  stage_skip(buf ^ 1, it + 1 - 8);
        }

        const unsigned short* pb = &ldsb[buf][0][n][kb];  // row 0, col n
        if (it < 8) {
            const unsigned short* ap =
                wfrag + ((((long)(b * 16 + mtg) * 8 + it) * 9) * 512 + lane * 16);
            __builtin_prefetch(ap + 9 * 512, 0, 2);       // next chunk's A block
            #pragma unroll
            for (int t = 0; t < 9; ++t) {
                const int dy = t / 3 - 1, dx = t % 3 - 1;
                v16bf a = *(const v16bf*)(ap + t * 512);
                union { uint4 q[2]; v16bf v; } u;
                const unsigned short* pt = pb + ((dy + 1) * 34 + (dx + 1)) * 32;
                u.q[0] = *(const uint4*)pt;        // channels kb..kb+7
                u.q[1] = *(const uint4*)(pt + 16); // channels kb+16..kb+23
                acc = __builtin_amdgcn_wmma_f32_16x16x32_bf16(
                    false, a, false, u.v, (short)0, acc, false, false);
            }
        } else if (FUSE) {                                // 1x1 skip chunk
            const unsigned short* ap =
                w1frag + (((long)mtg * 8 + (it - 8)) * 512 + lane * 16);
            v16bf a = *(const v16bf*)ap;
            union { uint4 q[2]; v16bf v; } u;
            u.q[0] = *(const uint4*)pb;
            u.q[1] = *(const uint4*)(pb + 16);
            acc = __builtin_amdgcn_wmma_f32_16x16x32_bf16(
                false, a, false, u.v, (short)0, acc, false, false);
        }

        wait_async0();       // own async copies to the other buffer done
        __syncthreads();     // slab visible to all waves
    }

    // Epilogue. C layout: VGPR r -> M = r + 8*(lane>=16), N = lane%16.
    // Lane holds 8 consecutive output channels at one pixel.
    const int xo = x0 + (nw << 4) + (lane & 15);
    const int o0 = mblk * 64 + mw * 16 + ((lane >> 4) << 3);
    if (FUSE) {
        #pragma unroll
        for (int r = 0; r < 8; ++r) {                    // NCHW f32 output
            int o = o0 + r;
            float v = acc[r] + bias[o];
            v = v >= 0.f ? v : LRELU_SLOPE * v;
            out_f[(((long)b * 256 + o) * 128 + y) * 128 + xo] = v;
        }
    } else {
        union { unsigned short h[8]; uint4 q; } pk;      // NHWC bf16, one b128
        #pragma unroll
        for (int r = 0; r < 8; ++r) {
            float v = acc[r] + bias[o0 + r];
            v = v >= 0.f ? v : LRELU_SLOPE * v;
            pk.h[r] = f2bf(v);
        }
        *(uint4*)(out_bf + (((long)(b * 128 + y) * 128 + xo) * 256 + o0)) = pk.q;
    }
}

// ---------------------------------------------------------------------------
extern "C" void kernel_launch(void* const* d_in, const int* in_sizes, int n_in,
                              void* d_out, int out_size, void* d_ws, size_t ws_size,
                              hipStream_t stream) {
    const float* x        = (const float*)d_in[0];
    const float* istyle   = (const float*)d_in[1];
    const float* w_style1 = (const float*)d_in[2];
    const float* b_style1 = (const float*)d_in[3];
    const float* w_style2 = (const float*)d_in[4];
    const float* b_style2 = (const float*)d_in[5];
    // d_in[6] w_noise1, d_in[8] w_noise2 unused (inoise == 0)
    const float* b_noise1 = (const float*)d_in[7];
    const float* b_noise2 = (const float*)d_in[9];
    const float* conv1_w  = (const float*)d_in[10];
    const float* conv2_w  = (const float*)d_in[11];
    const float* conv1x1_w= (const float*)d_in[12];
    float* out = (float*)d_out;

    char* ws = (char*)d_ws;
    size_t off = 0;
    auto take = [&](size_t bytes) {
        char* p = ws + off;
        off = (off + bytes + 255) & ~(size_t)255;
        return p;
    };
    float* s1  = (float*)take(4 * 256 * sizeof(float));
    float* s2  = (float*)take(4 * 256 * sizeof(float));
    float* dm1 = (float*)take(4 * 256 * sizeof(float));
    float* dm2 = (float*)take(4 * 256 * sizeof(float));
    unsigned short* xu   = (unsigned short*)take((size_t)4 * 128 * 128 * 256 * 2);
    unsigned short* act1 = (unsigned short*)take((size_t)4 * 128 * 128 * 256 * 2);
    unsigned short* wb1  = (unsigned short*)take((size_t)4 * 16 * 8 * 9 * 512 * 2);
    unsigned short* wb2  = (unsigned short*)take((size_t)4 * 16 * 8 * 9 * 512 * 2);
    unsigned short* w11  = (unsigned short*)take((size_t)16 * 8 * 512 * 2);
    (void)ws_size; (void)in_sizes; (void)n_in; (void)out_size;

    style_k<<<8, 256, 0, stream>>>(istyle, w_style1, b_style1, w_style2, b_style2, s1, s2);
    demod_k<<<8, 256, 0, stream>>>(conv1_w, conv2_w, s1, s2, dm1, dm2);
    upsample_k<<<65536, 256, 0, stream>>>(x, xu);

    swizzle_k<<<9216, 256, 0, stream>>>(conv1_w, s1, dm1, wb1, 1);
    swizzle_k<<<9216, 256, 0, stream>>>(conv2_w, s2, dm2, wb2, 1);
    swizzle_k<<<256,  256, 0, stream>>>(conv1x1_w, nullptr, nullptr, w11, 0);

    dim3 grid(512, 4, 4);   // (pixel tiles 128 rows x 4, M blocks, batch)
    conv_gemm_k<false><<<grid, 256, 0, stream>>>(xu, wb1, nullptr, nullptr,
                                                 b_noise1, act1, nullptr);
    conv_gemm_k<true><<<grid, 256, 0, stream>>>(act1, wb2, xu, w11,
                                                b_noise2, nullptr, out);
}
